// LSTMBlock_86457691668950
// MI455X (gfx1250) — matrix operand order repaired and checked by reading, not compiled, not run on hardware
//
#include <hip/hip_runtime.h>
#include <math.h>
#include <stdint.h>

// Problem constants (reference: B=32, T=512, D=512, U=512)
#define Bn 32
#define Tn 512
#define Dn 512
#define Un 512
#define Gn 2048          // 4*U
#define MROWS (Bn * Tn)  // 16384

typedef __attribute__((ext_vector_type(16))) __bf16 v16bf;
typedef __attribute__((ext_vector_type(8)))  float  v8f;
typedef __attribute__((ext_vector_type(4)))  unsigned int v4u;

// TDM descriptor groups (6-arg builtin signature: g0,g1,g2,g3,g4,cpol)
typedef __attribute__((ext_vector_type(4))) unsigned int u32x4;
typedef __attribute__((ext_vector_type(8))) int i32x8;
typedef __attribute__((ext_vector_type(4))) int i32x4;

typedef __attribute__((address_space(3))) unsigned short lds_us;

union BVec { v4u q[2]; v16bf v; };

__device__ __forceinline__ unsigned short f2bf(float f) {
  unsigned int u = __float_as_uint(f);
  u += 0x7FFFu + ((u >> 16) & 1u);   // round-to-nearest-even
  return (unsigned short)(u >> 16);
}

__device__ __forceinline__ float sigm(float x) {
  return 1.0f / (1.0f + __expf(-x));
}

// ---------------- workspace layout (bytes) ----------------
#define OFF_XG   ((size_t)0)                                   // f32 [2][MROWS][Gn]
#define OFF_WT   (OFF_XG + (size_t)2 * MROWS * Gn * 4)         // bf16 [2][Gn][Dn]  (W transposed)
#define OFF_UT   (OFF_WT + (size_t)2 * Gn * Dn * 2)            // bf16 [2][Gn][Un]  (U transposed)
#define OFF_XBF  (OFF_UT + (size_t)2 * Gn * Un * 2)            // bf16 [MROWS][Dn]
#define OFF_HBF  (OFF_XBF + (size_t)MROWS * Dn * 2)            // bf16 [2buf][2dir][Bn][Un]
#define OFF_CST  (OFF_HBF + (size_t)2 * 2 * Bn * Un * 2)       // f32  [2dir][Bn][Un]

// ---------------- init / conversion kernels ----------------

// src [K][N] f32 -> dst [N][K] bf16 (transpose + convert)
__global__ void k_cvt_t(const float* __restrict__ src, unsigned short* __restrict__ dst,
                        int K, int N) {
  int i = blockIdx.x * blockDim.x + threadIdx.x;
  if (i < K * N) {
    int nn = i / K, kk = i - nn * K;
    dst[i] = f2bf(src[kk * N + nn]);
  }
}

__global__ void k_cvt(const float* __restrict__ src, unsigned short* __restrict__ dst, int n) {
  int i = blockIdx.x * blockDim.x + threadIdx.x;
  if (i < n) dst[i] = f2bf(src[i]);
}

// h0 = c0 = z for both directions; write h into bf16 ping buffer 0
__global__ void k_state(const float* __restrict__ z, unsigned short* __restrict__ hbf,
                        float* __restrict__ cst) {
  int i = blockIdx.x * blockDim.x + threadIdx.x;
  if (i < Bn * Un) {
    unsigned short hb = f2bf(z[i]);
    hbf[0 * Bn * Un + i] = hb;            // buf0, dir0
    hbf[1 * Bn * Un + i] = hb;            // buf0, dir1
    cst[0 * Bn * Un + i] = z[i];
    cst[1 * Bn * Un + i] = z[i];
  }
}

// ---------------- big input-projection GEMM ----------------
// xg[dir][m][n] = sum_k x[m][k] * W[dir][k][n] + b[dir][n]
// grid = (MROWS/32, Gn/64, 2), block = 256 (8 waves; wave = one 16x16 tile)
__global__ __launch_bounds__(256) void k_inproj(const unsigned short* __restrict__ xbf,
                                                const unsigned short* __restrict__ wt,
                                                const float* __restrict__ bias_f,
                                                const float* __restrict__ bias_b,
                                                float* __restrict__ xg) {
  const int dir = blockIdx.z;
  const unsigned short* wtd = wt + (size_t)dir * Gn * Dn;
  const float* bias = dir ? bias_b : bias_f;
  float* xgo = xg + (size_t)dir * MROWS * Gn;

  const int mbase  = blockIdx.x * 32;
  const int nchunk = blockIdx.y * 64;
  const int w = threadIdx.x >> 5, lane = threadIdx.x & 31;
  const int mt = w & 1, nt = w >> 1;
  const int lo = lane & 15, hi = lane >> 4;
  const int col = nchunk + nt * 16 + lo;

  v8f acc0, acc1;
  const float bv = bias[col];
#pragma unroll
  for (int r = 0; r < 8; ++r) { acc0[r] = bv; acc1[r] = 0.0f; }

  const unsigned short* arow = xbf + (size_t)(mbase + mt * 16 + lo) * Dn;
  const unsigned short* brow = wtd + (size_t)col * Dn;
  const int khA = hi * 8, khB = hi * 8;

  // two independent accumulator chains to break WMMA RAW dependences
  for (int kk = 0; kk < Dn; kk += 64) {
    BVec a0, b0, a1, b1;
    a0.q[0] = *(const v4u*)(arow + kk + khA);
    a0.q[1] = *(const v4u*)(arow + kk + 16 + khA);
    b0.q[0] = *(const v4u*)(brow + kk + khB);
    b0.q[1] = *(const v4u*)(brow + kk + 16 + khB);
    a1.q[0] = *(const v4u*)(arow + kk + 32 + khA);
    a1.q[1] = *(const v4u*)(arow + kk + 48 + khA);
    b1.q[0] = *(const v4u*)(brow + kk + 32 + khB);
    b1.q[1] = *(const v4u*)(brow + kk + 48 + khB);
    acc0 = __builtin_amdgcn_wmma_f32_16x16x32_bf16(false, a0.v, false, b0.v,
                                                   (short)0, acc0, false, false);
    acc1 = __builtin_amdgcn_wmma_f32_16x16x32_bf16(false, a1.v, false, b1.v,
                                                   (short)0, acc1, false, false);
  }
#pragma unroll
  for (int r = 0; r < 8; ++r) {
    int row = mbase + mt * 16 + r + 8 * hi;
    xgo[(size_t)row * Gn + col] = acc0[r] + acc1[r];
  }
}

// ---------------- one recurrent timestep (both directions) ----------------
// blocks 0..31: fwd (t = s), blocks 32..63: bwd (t = T-1-s).
// Each block owns 16 hidden units, computes all 4 gates for all 32 batches,
// updates c/h, writes merged output. The full h matrix (32x512 bf16 = 32KB)
// is DMA'd into LDS once per block by the Tensor Data Mover.
__global__ __launch_bounds__(256) void k_step(const float* __restrict__ xg,
                                              const unsigned short* __restrict__ ut,
                                              unsigned short* __restrict__ hbf,
                                              float* __restrict__ cst,
                                              float* __restrict__ out, int s) {
  __shared__ unsigned short hsh[Bn * Un];   // 32 KB: staged h for this direction
  __shared__ float gs[8][8][32];

  const int dir = blockIdx.x >> 5;
  const int ub  = (blockIdx.x & 31) * 16;
  const int t   = dir ? (Tn - 1 - s) : s;
  const int w = threadIdx.x >> 5, lane = threadIdx.x & 31;
  const int mt = w & 1, gi = w >> 1;
  const int lo = lane & 15, hi = lane >> 4;
  const int cur = s & 1, nxt = cur ^ 1;
  const int n = gi * Un + ub + lo;   // column in [0,4U)

  const float* xgd = xg + (size_t)dir * MROWS * Gn;
  const unsigned short* brow = ut + ((size_t)dir * Gn + n) * Un;
  const int khA = hi * 8, khB = hi * 8;

  // --- TDM: issue one async tensor DMA of h[cur][dir] (32KB, contiguous) ---
  if (w == 0) {
    unsigned long long ga =
        (unsigned long long)(uintptr_t)(hbf + (size_t)(cur * 2 + dir) * Bn * Un);
    unsigned lds_off = (unsigned)(uintptr_t)(lds_us*)hsh;
    u32x4 g0;
    i32x8 g1;
    i32x4 gz4 = {0, 0, 0, 0};
    i32x8 gz8 = {0, 0, 0, 0, 0, 0, 0, 0};
    // D# group 0: count=1, lds_addr, global_addr, type=2
    g0[0] = 1u;
    g0[1] = lds_off;
    g0[2] = (unsigned)(ga & 0xFFFFFFFFu);
    g0[3] = (unsigned)((ga >> 32) & 0x01FFFFFFu) | 0x80000000u;
    // D# group 1: data_size=8B, 1-D tile of 4096 elements (= 32KB)
    g1[0] = 0x00030000;                 // workgroup_mask=0, data_size=3 (8B)
    g1[1] = (int)(0x1000u << 16);       // tensor_dim0[15:0] = 4096
    g1[2] = 0x00010000;                 // tensor_dim0[31:16]=0, tensor_dim1=1
    g1[3] = (int)(0x1000u << 16);       // tile_dim0 = 4096
    g1[4] = 0x00000001;                 // tile_dim1 = 1, tile_dim2 = 0
    g1[5] = 0x00001000;                 // tensor_dim0_stride = 4096
    g1[6] = 0;
    g1[7] = 0;
    __builtin_amdgcn_tensor_load_to_lds(g0, g1, gz4, gz4, gz8, 0);
  }

  // accumulator init from precomputed input projections (overlaps with TDM)
  v8f acc0, acc1;
#pragma unroll
  for (int r = 0; r < 8; ++r) {
    int br = mt * 16 + r + 8 * hi;
    acc0[r] = xgd[((size_t)br * Tn + t) * Gn + n];
    acc1[r] = 0.0f;
  }

  if (w == 0) __builtin_amdgcn_s_wait_tensorcnt(0);
  __syncthreads();

  const unsigned short* hrow = hsh + (size_t)(mt * 16 + lo) * Un;
  for (int kk = 0; kk < Un; kk += 64) {
    BVec a0, b0, a1, b1;
    a0.q[0] = *(const v4u*)(hrow + kk + khA);
    a0.q[1] = *(const v4u*)(hrow + kk + 16 + khA);
    b0.q[0] = *(const v4u*)(brow + kk + khB);
    b0.q[1] = *(const v4u*)(brow + kk + 16 + khB);
    a1.q[0] = *(const v4u*)(hrow + kk + 32 + khA);
    a1.q[1] = *(const v4u*)(hrow + kk + 48 + khA);
    b1.q[0] = *(const v4u*)(brow + kk + 32 + khB);
    b1.q[1] = *(const v4u*)(brow + kk + 48 + khB);
    acc0 = __builtin_amdgcn_wmma_f32_16x16x32_bf16(false, a0.v, false, b0.v,
                                                   (short)0, acc0, false, false);
    acc1 = __builtin_amdgcn_wmma_f32_16x16x32_bf16(false, a1.v, false, b1.v,
                                                   (short)0, acc1, false, false);
  }

#pragma unroll
  for (int r = 0; r < 8; ++r) gs[w][r][lane] = acc0[r] + acc1[r];
  __syncthreads();

  if (gi == 0) {  // waves 0 (mt=0) and 1 (mt=1) do the pointwise update
#pragma unroll
    for (int r = 0; r < 8; ++r) {
      float iv = sigm(gs[0 + mt][r][lane]);
      float fv = sigm(gs[2 + mt][r][lane]);
      float cc = tanhf(gs[4 + mt][r][lane]);
      float ov = sigm(gs[6 + mt][r][lane]);
      int br = mt * 16 + r + 8 * hi;
      int u  = ub + lo;
      size_t ci = ((size_t)dir * Bn + br) * Un + u;
      float cn = fv * cst[ci] + iv * cc;
      cst[ci] = cn;
      float h = ov * tanhf(cn);
      hbf[((size_t)(nxt * 2 + dir) * Bn + br) * Un + u] = f2bf(h);
      size_t oi = ((size_t)br * Tn + t) * Un + u;
      // first writer stores, second accumulates (deterministic via stream order)
      bool first = dir ? (t >= Tn / 2) : (t < Tn / 2);
      if (first) out[oi] = h;
      else       out[oi] += h;
    }
  }
}

// ---------------- host entry ----------------
extern "C" void kernel_launch(void* const* d_in, const int* in_sizes, int n_in,
                              void* d_out, int out_size, void* d_ws, size_t ws_size,
                              hipStream_t stream) {
  const float* x  = (const float*)d_in[0];
  const float* z  = (const float*)d_in[1];
  const float* Wf = (const float*)d_in[2];
  const float* Uf = (const float*)d_in[3];
  const float* bf = (const float*)d_in[4];
  const float* Wb = (const float*)d_in[5];
  const float* Ub = (const float*)d_in[6];
  const float* bb = (const float*)d_in[7];
  float* out = (float*)d_out;

  char* ws = (char*)d_ws;
  float*          xg  = (float*)(ws + OFF_XG);
  unsigned short* wt  = (unsigned short*)(ws + OFF_WT);
  unsigned short* ut  = (unsigned short*)(ws + OFF_UT);
  unsigned short* xbf = (unsigned short*)(ws + OFF_XBF);
  unsigned short* hbf = (unsigned short*)(ws + OFF_HBF);
  float*          cst = (float*)(ws + OFF_CST);

  const int cvtN = Dn * Gn;                       // 1,048,576 per matrix
  const int cvtB = (cvtN + 255) / 256;
  k_cvt_t<<<cvtB, 256, 0, stream>>>(Wf, wt,                 Dn, Gn);
  k_cvt_t<<<cvtB, 256, 0, stream>>>(Wb, wt + (size_t)Gn*Dn, Dn, Gn);
  k_cvt_t<<<cvtB, 256, 0, stream>>>(Uf, ut,                 Un, Gn);
  k_cvt_t<<<cvtB, 256, 0, stream>>>(Ub, ut + (size_t)Gn*Un, Un, Gn);

  const int xN = MROWS * Dn;
  k_cvt<<<(xN + 255) / 256, 256, 0, stream>>>(x, xbf, xN);
  k_state<<<(Bn * Un + 255) / 256, 256, 0, stream>>>(z, hbf, cst);

  // Input projections for all timesteps, both directions (one big WMMA GEMM).
  k_inproj<<<dim3(MROWS / 32, Gn / 64, 2), 256, 0, stream>>>(xbf, wt, bf, bb, xg);

  // Sequential recurrent scan: fwd t=s and bwd t=T-1-s fused per launch.
  for (int s = 0; s < Tn; ++s)
    k_step<<<64, 256, 0, stream>>>(xg, ut, hbf, cst, out, s);
}
